// PointNet2TreeSegmentor_31920196944630
// MI455X (gfx1250) — matrix operand order repaired and checked by reading, not compiled, not run on hardware
//
#include <hip/hip_runtime.h>
#include <cstdint>
#include <cstddef>

// ---------------------------------------------------------------------------
// PointNet++ (PointNet2TreeSegmentor) forward for gfx1250 (MI455X).
// All Linear layers run through v_wmma_f32_16x16x32_f16 (wave32 WMMA).
// One wave computes a 16x64 output tile (4 accumulators). The k-loop is
// software-pipelined and manually unrolled x2 with two fixed fragment
// register sets (ping/pong) -> no rotation copies, loads stay clause-grouped
// ahead of the WMMA groups that consume them.
// ---------------------------------------------------------------------------

typedef _Float16 h16v __attribute__((ext_vector_type(16)));
typedef _Float16 h8v  __attribute__((ext_vector_type(8)));
typedef float    f8v  __attribute__((ext_vector_type(8)));

static __device__ __forceinline__ size_t gidx() {
  return (size_t)blockIdx.x * blockDim.x + threadIdx.x;
}

// A fragment (16x32 f16): 16 contiguous K halves per lane (k = 16h + s)
static __device__ __forceinline__ h16v load_afrag(const _Float16* __restrict__ arow, int k) {
  h8v lo = *(const h8v*)(arow + k);
  h8v hi = *(const h8v*)(arow + k + 8);
  return __builtin_shufflevector(lo, hi, 0,1,2,3,4,5,6,7,8,9,10,11,12,13,14,15);
}

// ----------------------------- elementwise ---------------------------------
__global__ void k_zero_f32(float* p, int n) {
  size_t i = gidx(); if (i < (size_t)n) p[i] = 0.f;
}

__global__ void k_f32_to_f16(const float* __restrict__ in, _Float16* __restrict__ out, size_t n) {
  size_t i = gidx(); if (i < n) out[i] = (_Float16)in[i];
}

// transpose+pad weights: W[K,Nc] f32 (row-major) -> Wt[NcPad, ldW] f16, zero pad
__global__ void k_wprep(const float* __restrict__ W, int K, int Nc,
                        _Float16* __restrict__ Wt, int ldW, int NcPad) {
  size_t i = gidx(); size_t tot = (size_t)NcPad * ldW; if (i >= tot) return;
  int n = (int)(i / ldW), k = (int)(i % ldW);
  _Float16 v = (_Float16)0.f;
  if (n < Nc && k < K) v = (_Float16)W[(size_t)k * Nc + n];
  Wt[i] = v;
}

__global__ void k_copy_cols(const _Float16* __restrict__ src, int ldS,
                            _Float16* __restrict__ dst, int ldD, int dcol, int M, int C) {
  size_t i = gidx(); size_t tot = (size_t)M * C; if (i >= tot) return;
  int r = (int)(i / C), c = (int)(i % C);
  dst[(size_t)r * ldD + dcol + c] = src[(size_t)r * ldS + c];
}

__global__ void k_zero_cols(_Float16* dst, int ldD, int col0, int nc, int M) {
  size_t i = gidx(); size_t tot = (size_t)M * nc; if (i >= tot) return;
  int r = (int)(i / nc), c = (int)(i % nc);
  dst[(size_t)r * ldD + col0 + c] = (_Float16)0.f;
}

// --------------------------- WMMA GEMM core --------------------------------
// Out[M,Nc] = A[M,Kpad](f16) x Wt^T  (+bias). One wave -> 16x64 output tile.
// Optional per-channel sum / sum^2 accumulation (for BatchNorm stats).
__global__ __launch_bounds__(128)
void k_gemm_wmma(const _Float16* __restrict__ A, int ldA,
                 const _Float16* __restrict__ Wt, int ldW,
                 const float* __restrict__ bias,
                 _Float16* __restrict__ Out, int ldOut,
                 int Nc, int Ntiles, int Kpad,
                 float* __restrict__ sum, float* __restrict__ sumsq) {
  const int lane = threadIdx.x;                          // 0..31 (wave32)
  const int nt0  = (blockIdx.y * 4 + threadIdx.y) * 4;   // wave-uniform
  if (nt0 >= Ntiles) return;
  int nvalid = Ntiles - nt0; if (nvalid > 4) nvalid = 4;
  const int mbase = blockIdx.x * 16;
  const int l  = lane & 15;
  const bool hi = lane >= 16;

  const _Float16* arow = A + (size_t)(mbase + l) * ldA + (hi ? 16 : 0);
  // B fragments from transposed weights: lane row = ntile*16+l, 16 contiguous K halves.
  // Out-of-range tiles are clamped to the last valid tile (results discarded).
  const _Float16* brow[4];
  #pragma unroll
  for (int t = 0; t < 4; ++t) {
    int tt = nt0 + t; if (tt >= Ntiles) tt = Ntiles - 1;
    brow[t] = Wt + (size_t)(tt * 16 + l) * ldW + (hi ? 16 : 0);
  }

  f8v acc[4] = {{}, {}, {}, {}};
  h16v a0, a1, b0[4], b1[4];

  // prologue: load step 0 into set 0
  a0 = load_afrag(arow, 0);
  #pragma unroll
  for (int t = 0; t < 4; ++t) b0[t] = *(const h16v*)(brow[t]);

  int k = 32;
  // main loop: two k-steps per iteration, ping/pong register sets, no copies
  for (; k + 32 < Kpad; k += 64) {
    a1 = load_afrag(arow, k);
    #pragma unroll
    for (int t = 0; t < 4; ++t) b1[t] = *(const h16v*)(brow[t] + k);
    #pragma unroll
    for (int t = 0; t < 4; ++t)
      acc[t] = __builtin_amdgcn_wmma_f32_16x16x32_f16(
                 false, a0, false, b0[t], (short)0, acc[t], false, false);
    a0 = load_afrag(arow, k + 32);
    #pragma unroll
    for (int t = 0; t < 4; ++t) b0[t] = *(const h16v*)(brow[t] + k + 32);
    #pragma unroll
    for (int t = 0; t < 4; ++t)
      acc[t] = __builtin_amdgcn_wmma_f32_16x16x32_f16(
                 false, a1, false, b1[t], (short)0, acc[t], false, false);
  }
  if (k < Kpad) {          // one more load + two compute groups
    a1 = load_afrag(arow, k);
    #pragma unroll
    for (int t = 0; t < 4; ++t) b1[t] = *(const h16v*)(brow[t] + k);
    #pragma unroll
    for (int t = 0; t < 4; ++t)
      acc[t] = __builtin_amdgcn_wmma_f32_16x16x32_f16(
                 false, a0, false, b0[t], (short)0, acc[t], false, false);
    #pragma unroll
    for (int t = 0; t < 4; ++t)
      acc[t] = __builtin_amdgcn_wmma_f32_16x16x32_f16(
                 false, a1, false, b1[t], (short)0, acc[t], false, false);
  } else {                 // odd step count: just drain set 0
    #pragma unroll
    for (int t = 0; t < 4; ++t)
      acc[t] = __builtin_amdgcn_wmma_f32_16x16x32_f16(
                 false, a0, false, b0[t], (short)0, acc[t], false, false);
  }

  const int rbase = mbase + (hi ? 8 : 0);
  for (int t = 0; t < nvalid; ++t) {       // wave-uniform bound
    const int col = (nt0 + t) * 16 + l;
    float bv = (bias && col < Nc) ? bias[col] : 0.f;
    float s = 0.f, s2 = 0.f;
    #pragma unroll
    for (int r = 0; r < 8; ++r) {
      float v = acc[t][r] + bv;
      s += v; s2 += v * v;
      Out[(size_t)(rbase + r) * ldOut + col] = (_Float16)v;
    }
    if (sum) { atomicAdd(&sum[col], s); atomicAdd(&sumsq[col], s2); }
  }
}

// BatchNorm (biased var, eps=1e-5) + ReLU, in place on f16 activations
__global__ void k_bn_relu(_Float16* __restrict__ h, int ld, int M, int Nc,
                          const float* __restrict__ sum, const float* __restrict__ sumsq,
                          const float* __restrict__ gamma, const float* __restrict__ beta) {
  size_t i = gidx(); size_t tot = (size_t)M * Nc; if (i >= tot) return;
  int r = (int)(i / Nc), c = (int)(i % Nc);
  float invM = 1.f / (float)M;
  float mu  = sum[c] * invM;
  float var = sumsq[c] * invM - mu * mu;
  float sc  = gamma[c] * rsqrtf(var + 1e-5f);
  float sh  = beta[c] - mu * sc;
  float v = (float)h[(size_t)r * ld + c] * sc + sh;
  h[(size_t)r * ld + c] = (_Float16)fmaxf(v, 0.f);
}

// max over K=64 grouped neighbors
__global__ void k_maxagg(const _Float16* __restrict__ in, int ldIn, int Kn, int C,
                         int rowsOut, _Float16* __restrict__ out, int ldOut) {
  size_t i = gidx(); size_t tot = (size_t)rowsOut * C; if (i >= tot) return;
  int r = (int)(i / C), c = (int)(i % C);
  const _Float16* p = in + (size_t)r * Kn * ldIn + c;
  float m = -3.4e38f;
  for (int j = 0; j < Kn; ++j) m = fmaxf(m, (float)p[(size_t)j * ldIn]);
  out[(size_t)r * ldOut + c] = (_Float16)m;
}

// ----------------------------- FPS -----------------------------------------
__global__ void k_fps(const float* __restrict__ pos, int Np, int S, int* __restrict__ idx_out) {
  const int b = blockIdx.x, t = threadIdx.x;        // 256 threads
  const float* p = pos + (size_t)b * Np * 3;
  __shared__ float mind[2048];
  __shared__ float rv[256]; __shared__ int ri[256];
  __shared__ int cur;
  for (int i = t; i < Np; i += 256) mind[i] = 3.4e38f;
  if (t == 0) cur = 0;
  __syncthreads();
  for (int s = 0; s < S; ++s) {
    if (t == 0) idx_out[(size_t)b * S + s] = cur;
    __syncthreads();
    float cx = p[cur*3], cy = p[cur*3+1], cz = p[cur*3+2];
    float best = -1.f; int bi = 0x7fffffff;
    for (int i = t; i < Np; i += 256) {
      float dx = p[i*3]-cx, dy = p[i*3+1]-cy, dz = p[i*3+2]-cz;
      float d2 = dx*dx + dy*dy + dz*dz;
      float m = fminf(mind[i], d2); mind[i] = m;
      if (m > best || (m == best && i < bi)) { best = m; bi = i; }
    }
    rv[t] = best; ri[t] = bi; __syncthreads();
    for (int off = 128; off > 0; off >>= 1) {
      if (t < off) {
        if (rv[t+off] > rv[t] || (rv[t+off] == rv[t] && ri[t+off] < ri[t])) {
          rv[t] = rv[t+off]; ri[t] = ri[t+off];
        }
      }
      __syncthreads();
    }
    if (t == 0) cur = ri[0];
    __syncthreads();
  }
}

__global__ void k_gather_ctr(const float* __restrict__ pos, int Np,
                             const int* __restrict__ idx, int S, float* __restrict__ ctr, int Bn) {
  size_t i = gidx(); size_t tot = (size_t)Bn * S * 3; if (i >= tot) return;
  int d = (int)(i % 3); int cid = (int)(i / 3); int b = cid / S;
  ctr[i] = pos[((size_t)b * Np + idx[cid]) * 3 + d];
}

// radius-ball 64-nearest (tie-break lowest idx), pad with self (=nbr[0])
__global__ void k_group(const float* __restrict__ pos, int Np,
                        const float* __restrict__ ctr, int S, float r2,
                        int* __restrict__ nbr) {
  const int cid = blockIdx.x, t = threadIdx.x;      // 256 threads
  const int b = cid / S;
  const float* p = pos + (size_t)b * Np * 3;
  __shared__ float d2s[2048]; __shared__ unsigned char taken[2048];
  __shared__ float sv[256]; __shared__ int si[256];
  float cx = ctr[cid*3], cy = ctr[cid*3+1], cz = ctr[cid*3+2];
  for (int i = t; i < Np; i += 256) {
    float dx = p[i*3]-cx, dy = p[i*3+1]-cy, dz = p[i*3+2]-cz;
    d2s[i] = dx*dx + dy*dy + dz*dz; taken[i] = 0;
  }
  __syncthreads();
  int* out = nbr + (size_t)cid * 64;
  for (int k = 0; k < 64; ++k) {
    float best = 3.4e38f; int bi = 0x7fffffff;
    for (int i = t; i < Np; i += 256) {
      if (!taken[i]) {
        float d = d2s[i];
        if (d < best || (d == best && i < bi)) { best = d; bi = i; }
      }
    }
    sv[t] = best; si[t] = bi; __syncthreads();
    for (int off = 128; off > 0; off >>= 1) {
      if (t < off) {
        if (sv[t+off] < sv[t] || (sv[t+off] == sv[t] && si[t+off] < si[t])) {
          sv[t] = sv[t+off]; si[t] = si[t+off];
        }
      }
      __syncthreads();
    }
    if (t == 0) {
      if (sv[0] <= r2) { out[k] = si[0]; taken[si[0]] = 1; }
      else out[k] = out[0];
    }
    __syncthreads();
  }
}

// build grouped-MLP input rows: [x_j (Cx) | pos_j - ctr (3) | zero pad]
__global__ void k_group_input(const _Float16* __restrict__ xf, int Cx, int ldx,
                              const float* __restrict__ pos, int Np,
                              const int* __restrict__ nbr, const float* __restrict__ ctr,
                              int S, _Float16* __restrict__ A, int ldA, int rows) {
  size_t i = gidx(); size_t tot = (size_t)rows * ldA; if (i >= tot) return;
  int row = (int)(i / ldA), c = (int)(i % ldA);
  int j = row & 63; int cid = row >> 6; int b = cid / S;
  int nb = nbr[(size_t)cid * 64 + j];
  _Float16 v = (_Float16)0.f;
  if (c < Cx)           v = xf[((size_t)b * Np + nb) * ldx + c];
  else if (c < Cx + 3) { int d = c - Cx;
    v = (_Float16)(pos[((size_t)b * Np + nb) * 3 + d] - ctr[(size_t)cid * 3 + d]); }
  A[i] = v;
}

// k=3 inverse-d^2 weighted interpolation; writes f16 columns into dst
__global__ void k_knn_interp(const _Float16* __restrict__ xsrc, int ldS, int C, int Ns,
                             const float* __restrict__ psrc, const float* __restrict__ pdst,
                             int Nd, _Float16* __restrict__ dst, int ldD, int colOff) {
  const int did = blockIdx.x, t = threadIdx.x;      // 128 threads; did = b*Nd+i
  const int b = did / Nd;
  __shared__ float sv[128]; __shared__ int si[128];
  __shared__ int sel[3]; __shared__ float sd[3];
  float px = pdst[did*3], py = pdst[did*3+1], pz = pdst[did*3+2];
  const float* ps = psrc + (size_t)b * Ns * 3;
  for (int kk = 0; kk < 3; ++kk) {
    float best = 3.4e38f; int bi = 0x7fffffff;
    for (int i = t; i < Ns; i += 128) {
      bool skip = false;
      for (int q = 0; q < kk; ++q) if (sel[q] == i) skip = true;
      float dx = ps[i*3]-px, dy = ps[i*3+1]-py, dz = ps[i*3+2]-pz;
      float d2 = dx*dx + dy*dy + dz*dz;
      if (!skip && (d2 < best || (d2 == best && i < bi))) { best = d2; bi = i; }
    }
    sv[t] = best; si[t] = bi; __syncthreads();
    for (int off = 64; off > 0; off >>= 1) {
      if (t < off) {
        if (sv[t+off] < sv[t] || (sv[t+off] == sv[t] && si[t+off] < si[t])) {
          sv[t] = sv[t+off]; si[t] = si[t+off];
        }
      }
      __syncthreads();
    }
    if (t == 0) { sel[kk] = si[0]; sd[kk] = sv[0]; }
    __syncthreads();
  }
  float w0 = 1.f/fmaxf(sd[0],1e-16f), w1 = 1.f/fmaxf(sd[1],1e-16f), w2 = 1.f/fmaxf(sd[2],1e-16f);
  float inv = 1.f / (w0 + w1 + w2);
  const _Float16* r0 = xsrc + ((size_t)b*Ns + sel[0]) * ldS;
  const _Float16* r1 = xsrc + ((size_t)b*Ns + sel[1]) * ldS;
  const _Float16* r2 = xsrc + ((size_t)b*Ns + sel[2]) * ldS;
  for (int c = t; c < C; c += 128)
    dst[(size_t)did * ldD + colOff + c] =
      (_Float16)((w0*(float)r0[c] + w1*(float)r1[c] + w2*(float)r2[c]) * inv);
}

__global__ void k_final(const _Float16* __restrict__ in, int ld, float* __restrict__ out, int n) {
  size_t i = gidx(); if (i < (size_t)n) out[i] = (float)in[i * (size_t)ld];
}

// ------------------------------ host side ----------------------------------
struct Lyr {
  const float* W; int K; int Nc;
  const float* bias; const float* gamma; const float* beta;
  _Float16* Wt; int ldW; int NcPad;
};

static inline int r32(int v) { return (v + 31) & ~31; }
static inline int r16(int v) { return (v + 15) & ~15; }
static inline int cdiv(int a, int b) { return (a + b - 1) / b; }
static inline int cdivz(size_t a, int b) { return (int)((a + (size_t)b - 1) / (size_t)b); }

static const _Float16* run_chain(hipStream_t st, const _Float16* A0, int ldA0, int M,
                                 Lyr* L, int n, _Float16* slot0, _Float16* slot1,
                                 float* stats, _Float16* finalOut, int ldFinal) {
  const _Float16* in = A0; int ldIn = ldA0; int parity = 0;
  _Float16* out = nullptr;
  for (int i = 0; i < n; ++i) {
    bool last = (i == n - 1);
    out = (last && finalOut) ? finalOut : (parity ? slot1 : slot0);
    int ldOut = (last && finalOut) ? ldFinal : L[i].Nc;
    float *s = nullptr, *s2 = nullptr;
    if (!last) {
      s = stats; s2 = stats + L[i].NcPad;
      k_zero_f32<<<cdiv(2 * L[i].NcPad, 256), 256, 0, st>>>(stats, 2 * L[i].NcPad);
    }
    int Ntiles = L[i].NcPad / 16;
    dim3 g(M / 16, cdiv(Ntiles, 16)); dim3 bk(32, 4);   // 4 waves x 4 N-tiles each
    k_gemm_wmma<<<g, bk, 0, st>>>(in, ldIn, L[i].Wt, L[i].ldW, L[i].bias,
                                  out, ldOut, L[i].Nc, Ntiles, L[i].ldW, s, s2);
    if (!last) {
      size_t tot = (size_t)M * L[i].Nc;
      k_bn_relu<<<cdivz(tot, 256), 256, 0, st>>>(out, ldOut, M, L[i].Nc, s, s2,
                                                 L[i].gamma, L[i].beta);
    }
    in = out; ldIn = ldOut; parity ^= 1;
  }
  return out;
}

extern "C" void kernel_launch(void* const* d_in, const int* in_sizes, int n_in,
                              void* d_out, int out_size, void* d_ws, size_t ws_size,
                              hipStream_t stream) {
  (void)in_sizes; (void)n_in; (void)out_size; (void)ws_size;
  constexpr int Bn = 4, Np = 2048, S0 = 1024, S1 = 256, Kn = 64;
  constexpr int M0 = Bn * S0 * Kn;   // 262144
  constexpr int M1 = Bn * S1 * Kn;   // 65536
  constexpr int Mu0 = Bn * S0;       // 4096
  constexpr int Mu1 = Bn * Np;       // 8192

  const float* x   = (const float*)d_in[0];
  const float* pos = (const float*)d_in[1];
  auto F = [&](int i) { return (const float*)d_in[i]; };

  // jax.tree_leaves dict key order: W, b, beta, gamma
  Lyr sa0[3] = {
    {F(2), 6, 64,    F(3),  F(5),  F(4),  nullptr,0,0},
    {F(6), 64, 128,  F(7),  F(9),  F(8),  nullptr,0,0},
    {F(10),128,256,  F(11), nullptr,nullptr, nullptr,0,0}};
  Lyr sa1[3] = {
    {F(12),259,512,  F(13), F(15), F(14), nullptr,0,0},
    {F(16),512,1024, F(17), F(19), F(18), nullptr,0,0},
    {F(20),1024,2048,F(21), nullptr,nullptr, nullptr,0,0}};
  Lyr up0[3] = {
    {F(22),2307,2048,nullptr,F(24), F(23), nullptr,0,0},
    {F(25),2048,1024,nullptr,F(27), F(26), nullptr,0,0},
    {F(28),1024,512, nullptr,nullptr,nullptr, nullptr,0,0}};
  Lyr up1[3] = {
    {F(29),518,512,  nullptr,F(31), F(30), nullptr,0,0},
    {F(32),512,256,  nullptr,F(34), F(33), nullptr,0,0},
    {F(35),256,128,  nullptr,nullptr,nullptr, nullptr,0,0}};
  Lyr rg[3] = {
    {F(36),128,64,   nullptr,F(38), F(37), nullptr,0,0},
    {F(39),64,32,    nullptr,F(41), F(40), nullptr,0,0},
    {F(42),32,1,     nullptr,nullptr,nullptr, nullptr,0,0}};

  // --- workspace bump allocator (256B aligned) ---
  char* base = (char*)d_ws; size_t off = 0;
  auto alloc = [&](size_t bytes) -> void* {
    void* p = base + off; off = (off + bytes + 255) & ~(size_t)255; return p;
  };

  _Float16* slot0 = (_Float16*)alloc((size_t)M1 * 2048 * 2);  // 256MB arenas
  _Float16* slot1 = (_Float16*)alloc((size_t)M1 * 2048 * 2);
  _Float16* xh    = (_Float16*)alloc((size_t)Mu1 * 3 * 2);
  int*   idx0 = (int*)alloc((size_t)Bn * S0 * 4);
  int*   idx1 = (int*)alloc((size_t)Bn * S1 * 4);
  float* pos0 = (float*)alloc((size_t)Bn * S0 * 3 * 4);
  float* pos1 = (float*)alloc((size_t)Bn * S1 * 3 * 4);
  int*   nbr0 = (int*)alloc((size_t)Bn * S0 * Kn * 4);
  int*   nbr1 = (int*)alloc((size_t)Bn * S1 * Kn * 4);
  _Float16* A_sa0 = (_Float16*)alloc((size_t)M0 * 32 * 2);
  _Float16* A_sa1 = (_Float16*)alloc((size_t)M1 * 288 * 2);
  _Float16* A_up0 = (_Float16*)alloc((size_t)Mu0 * 2336 * 2);
  _Float16* A_up1 = (_Float16*)alloc((size_t)Mu1 * 544 * 2);
  _Float16* x0   = (_Float16*)alloc((size_t)Mu0 * 256 * 2);
  _Float16* x1   = (_Float16*)alloc((size_t)Bn * S1 * 2048 * 2);
  _Float16* x2p  = (_Float16*)alloc((size_t)Mu0 * 512 * 2);
  _Float16* x3p  = (_Float16*)alloc((size_t)Mu1 * 128 * 2);
  _Float16* oreg = (_Float16*)alloc((size_t)Mu1 * 16 * 2);
  float* stats   = (float*)alloc((size_t)2 * 2048 * 4);

  // --- weight prep (transpose + pad to f16) ---
  Lyr* chains[5] = {sa0, sa1, up0, up1, rg};
  for (int ci = 0; ci < 5; ++ci)
    for (int li = 0; li < 3; ++li) {
      Lyr& L = chains[ci][li];
      L.ldW = r32(L.K); L.NcPad = r16(L.Nc);
      L.Wt = (_Float16*)alloc((size_t)L.NcPad * L.ldW * 2);
      size_t tot = (size_t)L.NcPad * L.ldW;
      k_wprep<<<cdivz(tot, 256), 256, 0, stream>>>(L.W, L.K, L.Nc, L.Wt, L.ldW, L.NcPad);
    }

  // --- x -> f16 ---
  k_f32_to_f16<<<cdivz((size_t)Mu1 * 3, 256), 256, 0, stream>>>(x, xh, (size_t)Mu1 * 3);

  // --- SA0: fps -> group -> grouped MLP -> max-agg ---
  k_fps<<<Bn, 256, 0, stream>>>(pos, Np, S0, idx0);
  k_gather_ctr<<<cdivz((size_t)Bn*S0*3,256),256,0,stream>>>(pos, Np, idx0, S0, pos0, Bn);
  k_group<<<Bn*S0, 256, 0, stream>>>(pos, Np, pos0, S0, 0.2f*0.2f, nbr0);
  k_group_input<<<cdivz((size_t)M0*32,256),256,0,stream>>>(xh, 3, 3, pos, Np, nbr0, pos0,
                                                           S0, A_sa0, 32, M0);
  const _Float16* h3 = run_chain(stream, A_sa0, 32, M0, sa0, 3, slot0, slot1, stats, nullptr, 0);
  k_maxagg<<<cdivz((size_t)Mu0*256,256),256,0,stream>>>(h3, 256, Kn, 256, Mu0, x0, 256);

  // --- SA1 ---
  k_fps<<<Bn, 256, 0, stream>>>(pos0, S0, S1, idx1);
  k_gather_ctr<<<cdivz((size_t)Bn*S1*3,256),256,0,stream>>>(pos0, S0, idx1, S1, pos1, Bn);
  k_group<<<Bn*S1, 256, 0, stream>>>(pos0, S0, pos1, S1, 0.4f*0.4f, nbr1);
  k_group_input<<<cdivz((size_t)M1*288,256),256,0,stream>>>(x0, 256, 256, pos0, S0, nbr1,
                                                            pos1, S1, A_sa1, 288, M1);
  h3 = run_chain(stream, A_sa1, 288, M1, sa1, 3, slot0, slot1, stats, nullptr, 0);
  k_maxagg<<<cdivz((size_t)Bn*S1*2048,256),256,0,stream>>>(h3, 2048, Kn, 2048, Bn*S1, x1, 2048);

  // --- up0: knn(x1 -> pos0) ++ x0 ++ zeros3, unit MLP ---
  k_knn_interp<<<Bn*S0, 128, 0, stream>>>(x1, 2048, 2048, S1, pos1, pos0, S0, A_up0, 2336, 0);
  k_copy_cols<<<cdivz((size_t)Mu0*256,256),256,0,stream>>>(x0, 256, A_up0, 2336, 2048, Mu0, 256);
  k_zero_cols<<<cdivz((size_t)Mu0*32,256),256,0,stream>>>(A_up0, 2336, 2304, 32, Mu0);
  run_chain(stream, A_up0, 2336, Mu0, up0, 3, slot0, slot1, stats, x2p, 512);

  // --- up1: knn(x2 -> pos) ++ x ++ zeros3, unit MLP ---
  k_knn_interp<<<Bn*Np, 128, 0, stream>>>(x2p, 512, 512, S0, pos0, pos, Np, A_up1, 544, 0);
  k_copy_cols<<<cdivz((size_t)Mu1*3,256),256,0,stream>>>(xh, 3, A_up1, 544, 512, Mu1, 3);
  k_zero_cols<<<cdivz((size_t)Mu1*29,256),256,0,stream>>>(A_up1, 544, 515, 29, Mu1);
  run_chain(stream, A_up1, 544, Mu1, up1, 3, slot0, slot1, stats, x3p, 128);

  // --- regression head ---
  run_chain(stream, x3p, 128, Mu1, rg, 3, slot0, slot1, stats, oreg, 16);

  k_final<<<cdiv(Mu1, 256), 256, 0, stream>>>(oreg, 16, (float*)d_out, Mu1);
}
